// Transformer_2491081031820
// MI455X (gfx1250) — matrix operand order repaired and checked by reading, not compiled, not run on hardware
//
#include <hip/hip_runtime.h>
#include <hip/hip_bf16.h>

// ---------------------------------------------------------------------------
// Shapes (fixed by the reference)
// ---------------------------------------------------------------------------
#define S_ 1024
#define B_ 4
#define D_ 1024
#define H_ 4096
#define V_ 32000
#define T_ (S_ * B_)   // 4096 tokens

typedef __attribute__((ext_vector_type(16))) _Float16 v16h;
typedef __attribute__((ext_vector_type(8)))  float    v8f;

// ---------------------------------------------------------------------------
// f32 -> f16 cast (weights), grid-stride
// ---------------------------------------------------------------------------
__global__ void cvt_f16_kernel(const float* __restrict__ src,
                               _Float16* __restrict__ dst, long n) {
    long i = (long)blockIdx.x * blockDim.x + threadIdx.x;
    long stride = (long)gridDim.x * blockDim.x;
    for (; i < n; i += stride) dst[i] = (_Float16)src[i];
}

// ---------------------------------------------------------------------------
// Embedding gather: emb[t, d] = table[ids[t], d]  (t = s*B + b), f32 + f16
// ---------------------------------------------------------------------------
__global__ void embed_kernel(const int* __restrict__ ids,
                             const float* __restrict__ table,
                             float* __restrict__ ef, _Float16* __restrict__ eh) {
    long i = (long)blockIdx.x * blockDim.x + threadIdx.x;
    long stride = (long)gridDim.x * blockDim.x;
    const long n = (long)T_ * D_;
    for (; i < n; i += stride) {
        long t = i / D_;
        int  d = (int)(i - t * D_);
        float v = table[(long)ids[t] * D_ + d];
        ef[i] = v;
        eh[i] = (_Float16)v;
    }
}

// ---------------------------------------------------------------------------
// V transpose: VT[b][d][s] = Vh[(s*B+b)*D + d]   (so attn @ V becomes x @ W^T)
// ---------------------------------------------------------------------------
__global__ void transposeV_kernel(const _Float16* __restrict__ Vh,
                                  _Float16* __restrict__ VT) {
    long i = (long)blockIdx.x * blockDim.x + threadIdx.x;
    long stride = (long)gridDim.x * blockDim.x;
    const long n = (long)B_ * D_ * S_;
    for (; i < n; i += stride) {
        int  s = (int)(i % S_);
        long r = i / S_;
        int  d = (int)(r % D_);
        int  b = (int)(r / D_);
        VT[i] = Vh[((long)s * B_ + b) * D_ + d];
    }
}

// ---------------------------------------------------------------------------
// WMMA GEMM:  C[m,n] = act( scale * sum_k A[m,k]*W[n,k] + bias[n] )
//   Block tile 64x256, K-step 32, 8 wave32s in a 2x4 grid, wave tile 32x64
//   = 2x4 v_wmma_f32_16x16x32_f16 accumulators.
//   Double-buffered LDS tiles filled by async global->LDS DMA
//   (global_load_async_to_lds_b128, ASYNCcnt): tile k+1 streams into LDS
//   while tile k is consumed by the WMMA pipe. No VGPR staging.
//   All M,N,K must divide 64/256/32 (true for every GEMM in this net).
// ---------------------------------------------------------------------------
#define BLK_M 64
#define BLK_N 256
#define BLK_K 32
#define LDS_PAD 8   // halves; row stride 40 halves = 80B (16B-aligned, conflict-avoiding)

__launch_bounds__(256)
__global__ void gemm_wmma_kernel(const _Float16* __restrict__ A, long lda, long sAb,
                                 const _Float16* __restrict__ W, long ldw, long sWb,
                                 float* __restrict__ Cf, _Float16* __restrict__ Ch,
                                 long ldc, long sCb,
                                 const float* __restrict__ bias,
                                 int K, float scale, int relu) {
    __shared__ __align__(16) _Float16 sA[2][BLK_M][BLK_K + LDS_PAD];
    __shared__ __align__(16) _Float16 sB[2][BLK_N][BLK_K + LDS_PAD];

    const int tid  = threadIdx.x;
    const int wave = tid >> 5;
    const int lane = tid & 31;
    const int wm   = wave >> 2;      // 0..1
    const int wn   = wave & 3;       // 0..3
    const int hl   = lane & 15;
    const int hi   = lane >> 4;

    const int bz = blockIdx.z;
    const _Float16* Ab = A + (long)bz * sAb;
    const _Float16* Wb = W + (long)bz * sWb;

    const long m0 = (long)blockIdx.y * BLK_M;
    const long n0 = (long)blockIdx.x * BLK_N;

    v8f acc[2][4];
#pragma unroll
    for (int i = 0; i < 2; ++i)
#pragma unroll
        for (int j = 0; j < 4; ++j)
#pragma unroll
            for (int r = 0; r < 8; ++r) acc[i][j][r] = 0.0f;

    // Cooperative tile assignment:
    //   A tile (64 rows x 32 halves): row = tid/4, 8-half chunk = tid%4 -> 1 B128
    //   B tile (256 rows x 32 halves): row = tid, 4 consecutive B128 chunks
    const int arow = tid >> 2;
    const int achk = (tid & 3) * 8;

    const _Float16* gA = Ab + (m0 + arow) * lda + achk;  // += k per tile
    const _Float16* gB = Wb + (n0 + tid) * ldw;          // += k per tile

    // LDS byte offsets (flat addr low 32 bits == group-segment offset)
    unsigned ldsA[2], ldsB[2];
    ldsA[0] = (unsigned)(uintptr_t)&sA[0][arow][achk];
    ldsA[1] = (unsigned)(uintptr_t)&sA[1][arow][achk];
    ldsB[0] = (unsigned)(uintptr_t)&sB[0][tid][0];
    ldsB[1] = (unsigned)(uintptr_t)&sB[1][tid][0];

    // Issue one tile's async DMA: 5 x global_load_async_to_lds_b128 / thread.
    auto issue_tile = [&](int kk, int buf) {
        unsigned long long a = (unsigned long long)(uintptr_t)(gA + kk);
        unsigned long long b = (unsigned long long)(uintptr_t)(gB + kk);
        asm volatile("global_load_async_to_lds_b128 %0, %1, off"
                     :: "v"(ldsA[buf]), "v"(a) : "memory");
        asm volatile("global_load_async_to_lds_b128 %0, %1, off"
                     :: "v"(ldsB[buf]), "v"(b) : "memory");
        asm volatile("global_load_async_to_lds_b128 %0, %1, off offset:16"
                     :: "v"(ldsB[buf]), "v"(b) : "memory");
        asm volatile("global_load_async_to_lds_b128 %0, %1, off offset:32"
                     :: "v"(ldsB[buf]), "v"(b) : "memory");
        asm volatile("global_load_async_to_lds_b128 %0, %1, off offset:48"
                     :: "v"(ldsB[buf]), "v"(b) : "memory");
    };

    const int nK = K / BLK_K;
    issue_tile(0, 0);                       // prologue: tile 0 -> buffer 0

    for (int kt = 0; kt < nK; ++kt) {
        const int cur = kt & 1;
        if (kt + 1 < nK) {
            // Previous end-of-body barrier guarantees buffer cur^1 is free.
            issue_tile((kt + 1) * BLK_K, cur ^ 1);
            // Async loads complete in order: <=5 outstanding => tile kt landed.
            asm volatile("s_wait_asynccnt 0x5" ::: "memory");
        } else {
            asm volatile("s_wait_asynccnt 0x0" ::: "memory");
        }
        __syncthreads();                    // all waves' DMA for tile kt visible

        const _Float16 (*cA)[BLK_K + LDS_PAD] = sA[cur];
        const _Float16 (*cB)[BLK_K + LDS_PAD] = sB[cur];

        // A fragments (16x32 f16 layout): lane<16 -> M=hl, K 0..7 & 16..23;
        //                                 lane>=16 -> K 8..15 & 24..31
        v16h afrag[2];
#pragma unroll
        for (int i = 0; i < 2; ++i) {
            const int r = wm * 32 + i * 16 + hl;
            union { uint4 q[2]; v16h h; } u;
            u.q[0] = *(const uint4*)&cA[r][hi * 8];
            u.q[1] = *(const uint4*)&cA[r][16 + hi * 8];
            afrag[i] = u.h;
        }
        // B fragments (32x16): lane<16 -> col=hl, K 0..7 & 16..23; lane>=16 -> K 8..15 & 24..31
        v16h bfrag[4];
#pragma unroll
        for (int j = 0; j < 4; ++j) {
            const int c = wn * 64 + j * 16 + hl;
            union { uint4 q[2]; v16h h; } u;
            u.q[0] = *(const uint4*)&cB[c][hi * 8];
            u.q[1] = *(const uint4*)&cB[c][16 + hi * 8];
            bfrag[j] = u.h;
        }
#pragma unroll
        for (int i = 0; i < 2; ++i)
#pragma unroll
            for (int j = 0; j < 4; ++j)
                acc[i][j] = __builtin_amdgcn_wmma_f32_16x16x32_f16(
                    false, afrag[i], false, bfrag[j],
                    (short)0, acc[i][j], false, false);

        __syncthreads();                    // reads done before buffer reuse
    }

    // Epilogue: C/D layout -> VGPR r = row (base + hi*8 + r), col = base + hl
#pragma unroll
    for (int i = 0; i < 2; ++i) {
#pragma unroll
        for (int j = 0; j < 4; ++j) {
            const long row0 = m0 + wm * 32 + i * 16 + hi * 8;
            const long col  = n0 + wn * 64 + j * 16 + hl;
            const float bv  = bias ? bias[col] : 0.0f;
#pragma unroll
            for (int r = 0; r < 8; ++r) {
                float v = acc[i][j][r] * scale + bv;
                if (relu) v = fmaxf(v, 0.0f);
                const long off = (long)bz * sCb + (row0 + r) * ldc + col;
                if (Cf) Cf[off] = v;
                if (Ch) Ch[off] = (_Float16)v;
            }
        }
    }
}

// ---------------------------------------------------------------------------
// Row softmax over scores[row, 0..S) -> f16 attn
// ---------------------------------------------------------------------------
__global__ void softmax_kernel(const float* __restrict__ scores,
                               _Float16* __restrict__ attn) {
    __shared__ float red[256];
    const long base = (long)blockIdx.x * S_;
    const int t = threadIdx.x;
    float vmax = -3.4e38f;
    for (int i = t; i < S_; i += 256) vmax = fmaxf(vmax, scores[base + i]);
    red[t] = vmax; __syncthreads();
    for (int s = 128; s > 0; s >>= 1) {
        if (t < s) red[t] = fmaxf(red[t], red[t + s]);
        __syncthreads();
    }
    vmax = red[0]; __syncthreads();
    float sum = 0.0f;
    for (int i = t; i < S_; i += 256) sum += __expf(scores[base + i] - vmax);
    red[t] = sum; __syncthreads();
    for (int s = 128; s > 0; s >>= 1) {
        if (t < s) red[t] += red[t + s];
        __syncthreads();
    }
    const float inv = 1.0f / red[0];
    for (int i = t; i < S_; i += 256)
        attn[base + i] = (_Float16)(__expf(scores[base + i] - vmax) * inv);
}

// ---------------------------------------------------------------------------
// Fused residual add + LayerNorm over D; one block per token row.
// out = (x+y - mean)/sqrt(var+eps)*g + be ; writes f32 (optional) + f16
// ---------------------------------------------------------------------------
__global__ void add_ln_kernel(const float* __restrict__ x,
                              const float* __restrict__ y,
                              const float* __restrict__ g,
                              const float* __restrict__ be,
                              float* __restrict__ outf,
                              _Float16* __restrict__ outh) {
    __shared__ float red[256];
    const long base = (long)blockIdx.x * D_;
    const int t = threadIdx.x;
    float s1 = 0.0f, s2 = 0.0f;
    for (int i = t; i < D_; i += 256) {
        float v = x[base + i] + y[base + i];
        s1 += v; s2 += v * v;
    }
    red[t] = s1; __syncthreads();
    for (int s = 128; s > 0; s >>= 1) { if (t < s) red[t] += red[t + s]; __syncthreads(); }
    s1 = red[0]; __syncthreads();
    red[t] = s2; __syncthreads();
    for (int s = 128; s > 0; s >>= 1) { if (t < s) red[t] += red[t + s]; __syncthreads(); }
    s2 = red[0];
    const float mean = s1 * (1.0f / D_);
    const float var  = s2 * (1.0f / D_) - mean * mean;
    const float rstd = rsqrtf(var + 1e-5f);
    for (int i = t; i < D_; i += 256) {
        float v = x[base + i] + y[base + i];
        float o = (v - mean) * rstd * g[i] + be[i];
        if (outf) outf[base + i] = o;
        outh[base + i] = (_Float16)o;
    }
}

// ---------------------------------------------------------------------------
// Host-side orchestration
// ---------------------------------------------------------------------------
static inline void launch_gemm(const _Float16* A, long lda, long sAb,
                               const _Float16* W, long ldw, long sWb,
                               float* Cf, _Float16* Ch, long ldc, long sCb,
                               const float* bias, int M, int N, int K,
                               float scale, int relu, int batch, hipStream_t st) {
    dim3 grid((unsigned)(N / BLK_N), (unsigned)(M / BLK_M), (unsigned)batch);
    gemm_wmma_kernel<<<grid, 256, 0, st>>>(A, lda, sAb, W, ldw, sWb,
                                           Cf, Ch, ldc, sCb, bias, K, scale, relu);
}

static inline void launch_cvt(const float* s, _Float16* d, long n, hipStream_t st) {
    long blocks = (n + 255) / 256;
    if (blocks > 4096) blocks = 4096;
    cvt_f16_kernel<<<(unsigned)blocks, 256, 0, st>>>(s, d, n);
}

extern "C" void kernel_launch(void* const* d_in, const int* in_sizes, int n_in,
                              void* d_out, int out_size, void* d_ws, size_t ws_size,
                              hipStream_t stream) {
    (void)in_sizes; (void)n_in; (void)out_size; (void)ws_size;
    const int*   ids   = (const int*)  d_in[0];
    const float* table = (const float*)d_in[1];
    const float* WQ    = (const float*)d_in[2];
    const float* WK    = (const float*)d_in[3];
    const float* WV    = (const float*)d_in[4];
    const float* Wo    = (const float*)d_in[5];
    const float* bo    = (const float*)d_in[6];
    const float* W1    = (const float*)d_in[7];
    const float* b1    = (const float*)d_in[8];
    const float* W2    = (const float*)d_in[9];
    const float* b2    = (const float*)d_in[10];
    const float* g1    = (const float*)d_in[11];
    const float* be1   = (const float*)d_in[12];
    const float* g2    = (const float*)d_in[13];
    const float* be2   = (const float*)d_in[14];
    const float* Wd    = (const float*)d_in[15];
    const float* bd    = (const float*)d_in[16];
    float* logits = (float*)d_out;

    // Carve scratch (256-B aligned slices)
    char* ws = (char*)d_ws;
    size_t off = 0;
    auto alloc = [&](size_t bytes) -> void* {
        void* p = ws + off;
        off += (bytes + 255) & ~(size_t)255;
        return p;
    };
    _Float16* WQh = (_Float16*)alloc((size_t)D_ * D_ * 2);
    _Float16* WKh = (_Float16*)alloc((size_t)D_ * D_ * 2);
    _Float16* WVh = (_Float16*)alloc((size_t)D_ * D_ * 2);
    _Float16* Woh = (_Float16*)alloc((size_t)D_ * D_ * 2);
    _Float16* W1h = (_Float16*)alloc((size_t)H_ * D_ * 2);
    _Float16* W2h = (_Float16*)alloc((size_t)D_ * H_ * 2);
    _Float16* Wdh = (_Float16*)alloc((size_t)V_ * D_ * 2);
    float*    embf = (float*)   alloc((size_t)T_ * D_ * 4);
    _Float16* embh = (_Float16*)alloc((size_t)T_ * D_ * 2);
    _Float16* Qh   = (_Float16*)alloc((size_t)T_ * D_ * 2);
    _Float16* Kh   = (_Float16*)alloc((size_t)T_ * D_ * 2);
    _Float16* Vh   = (_Float16*)alloc((size_t)T_ * D_ * 2);
    _Float16* VTh  = (_Float16*)alloc((size_t)B_ * D_ * S_ * 2);
    float*    scf  = (float*)   alloc((size_t)B_ * S_ * S_ * 4);
    _Float16* atth = (_Float16*)alloc((size_t)B_ * S_ * S_ * 2);
    _Float16* zh   = (_Float16*)alloc((size_t)T_ * D_ * 2);
    float*    zof  = (float*)   alloc((size_t)T_ * D_ * 4);
    float*    x1f  = (float*)   alloc((size_t)T_ * D_ * 4);
    _Float16* x1h  = (_Float16*)alloc((size_t)T_ * D_ * 2);
    _Float16* hh   = (_Float16*)alloc((size_t)T_ * H_ * 2);
    float*    f2f  = (float*)   alloc((size_t)T_ * D_ * 4);
    _Float16* x2h  = (_Float16*)alloc((size_t)T_ * D_ * 2);

    // 1) weight casts f32 -> f16
    launch_cvt(WQ, WQh, (long)D_ * D_, stream);
    launch_cvt(WK, WKh, (long)D_ * D_, stream);
    launch_cvt(WV, WVh, (long)D_ * D_, stream);
    launch_cvt(Wo, Woh, (long)D_ * D_, stream);
    launch_cvt(W1, W1h, (long)H_ * D_, stream);
    launch_cvt(W2, W2h, (long)D_ * H_, stream);
    launch_cvt(Wd, Wdh, (long)V_ * D_, stream);

    // 2) embedding gather
    embed_kernel<<<4096, 256, 0, stream>>>(ids, table, embf, embh);

    // 3) Q/K/V projections  [T,D] x [D,D]^T
    launch_gemm(embh, D_, 0, WQh, D_, 0, nullptr, Qh, D_, 0, nullptr,
                T_, D_, D_, 1.0f, 0, 1, stream);
    launch_gemm(embh, D_, 0, WKh, D_, 0, nullptr, Kh, D_, 0, nullptr,
                T_, D_, D_, 1.0f, 0, 1, stream);
    launch_gemm(embh, D_, 0, WVh, D_, 0, nullptr, Vh, D_, 0, nullptr,
                T_, D_, D_, 1.0f, 0, 1, stream);

    // 4) V transpose per batch
    transposeV_kernel<<<4096, 256, 0, stream>>>(Vh, VTh);

    // 5) scores[b] = Q_b K_b^T / sqrt(D)  (batched via grid.z, strided rows)
    launch_gemm(Qh, (long)B_ * D_, D_,  Kh, (long)B_ * D_, D_,
                scf, nullptr, S_, (long)S_ * S_, nullptr,
                S_, S_, D_, 0.03125f, 0, B_, stream);

    // 6) softmax rows -> f16 attention
    softmax_kernel<<<B_ * S_, 256, 0, stream>>>(scf, atth);

    // 7) z[b] = attn_b @ V_b  ==  attn_b @ (VT_b)^T ; write token-major f16
    launch_gemm(atth, S_, (long)S_ * S_,  VTh, S_, (long)D_ * S_,
                nullptr, zh, (long)B_ * D_, D_, nullptr,
                S_, D_, S_, 1.0f, 0, B_, stream);

    // 8) attention out projection + bo
    launch_gemm(zh, D_, 0, Woh, D_, 0, zof, nullptr, D_, 0, bo,
                T_, D_, D_, 1.0f, 0, 1, stream);

    // 9) x1 = LN(emb + z@Wo)
    add_ln_kernel<<<T_, 256, 0, stream>>>(embf, zof, g1, be1, x1f, x1h);

    // 10) FFN up + ReLU (fused)
    launch_gemm(x1h, D_, 0, W1h, D_, 0, nullptr, hh, H_, 0, b1,
                T_, H_, D_, 1.0f, 1, 1, stream);

    // 11) FFN down
    launch_gemm(hh, H_, 0, W2h, H_, 0, f2f, nullptr, D_, 0, b2,
                T_, D_, H_, 1.0f, 0, 1, stream);

    // 12) x2 = LN(x1 + ffn)
    add_ln_kernel<<<T_, 256, 0, stream>>>(x1f, f2f, g2, be2, nullptr, x2h);

    // 13) decoder head: logits = x2 @ Wd^T + bd  (the 268-GFLOP GEMM)
    launch_gemm(x2h, D_, 0, Wdh, D_, 0, logits, nullptr, V_, 0, bd,
                T_, V_, D_, 1.0f, 0, 1, stream);
}
